// SwinTransformerBlock_40398462386528
// MI455X (gfx1250) — matrix operand order repaired and checked by reading, not compiled, not run on hardware
//
#include <hip/hip_runtime.h>
#include <cstdint>

#define DIM    96
#define HEADS  3
#define HD     32
#define WINSZ  7
#define NTOK   49          // tokens per window
#define MP     64          // padded window rows for 16x16 tiles
#define HGT    224
#define WID    224
#define BATCH  8
#define HWCNT  (HGT*WID)   // 50176
#define NWIN   (BATCH*32*32) // 8192 windows
#define HID    384
#define C3     (3*DIM)     // 288

typedef __attribute__((ext_vector_type(16))) _Float16 v16h;
typedef __attribute__((ext_vector_type(8)))  float    v8f;
typedef __attribute__((ext_vector_type(4)))  int      v4i;

#if __has_builtin(__builtin_amdgcn_global_load_async_to_lds_b128)
#define HAVE_ASYNC_LDS 1
#endif

union F16Frag {
  v16h     h;
  uint32_t u[8];
  uint4    q[2];
};

// A-matrix 16x32 f16 fragment (M rows striped in lanes, K in VGPRs).
// Lane l: m = m0 + (l&15); hf = l>>4.
// VGPRs 0..3 hold K = k0+hf*8 .. +7 (16 contiguous bytes); VGPRs 4..7 hold
// K = k0+16+hf*8 .. +7. Both halves are 16B-aligned -> 2x ds_load_b128.
__device__ __forceinline__ void load_a(F16Frag& f, const _Float16* buf, int ldk,
                                       int m0, int k0, int lane) {
  const int m  = m0 + (lane & 15);
  const int hf = lane >> 4;
  const _Float16* row = buf + (size_t)m * ldk + hf * 8 + k0;
  f.q[0] = *(const uint4*)(row);
  f.q[1] = *(const uint4*)(row + 16);
}

// B-matrix 32x16 f16 fragment from N-major storage buf[n][k] (row stride ldk halves).
// Lane l: n = n0 + (l&15); VGPRs 0..3: K = k0+(l>>4)*8 .. +7; VGPRs 4..7: K = k0+16+(l>>4)*8 .. +7.
__device__ __forceinline__ void load_b(F16Frag& f, const _Float16* buf, int ldk,
                                       int k0, int n0, int lane) {
  const int n  = n0 + (lane & 15);
  const int hf = lane >> 4;
  const uint4* p = (const uint4*)(buf + (size_t)n * ldk + k0 + hf * 8);
  f.q[0] = p[0];
  f.q[1] = p[2];
}

__device__ __forceinline__ v8f wmma16(const F16Frag& a, const F16Frag& b, v8f c) {
  return __builtin_amdgcn_wmma_f32_16x16x32_f16(false, a.h, false, b.h,
                                                (short)0, c, false, false);
}

// 4-lane (within-wave) reductions: lanes 4r..4r+3 cooperate on one row.
__device__ __forceinline__ float red4_sum(float v) {
  v += __shfl_xor(v, 1, 32);
  v += __shfl_xor(v, 2, 32);
  return v;
}
__device__ __forceinline__ float red4_max(float v) {
  v = fmaxf(v, __shfl_xor(v, 1, 32));
  v = fmaxf(v, __shfl_xor(v, 2, 32));
  return v;
}

// ---------------- weight transpose/convert to f16, N-major ----------------
__global__ void prep_kernel(const float* __restrict__ qkv_w,
                            const float* __restrict__ proj_w,
                            const float* __restrict__ fc1_w,
                            const float* __restrict__ fc2_w,
                            _Float16* __restrict__ qkvT,
                            _Float16* __restrict__ projT,
                            _Float16* __restrict__ fc1T,
                            _Float16* __restrict__ fc2T) {
  const int i = blockIdx.x * blockDim.x + threadIdx.x;
  const int stride = gridDim.x * blockDim.x;
  for (int idx = i; idx < C3 * DIM; idx += stride) {          // qkvT[288][96]
    int n = idx / DIM, k = idx - n * DIM;
    qkvT[idx] = (_Float16)qkv_w[k * C3 + n];
  }
  for (int idx = i; idx < DIM * DIM; idx += stride) {         // projT[96][96]
    int n = idx / DIM, k = idx - n * DIM;
    projT[idx] = (_Float16)proj_w[k * DIM + n];
  }
  for (int idx = i; idx < HID * DIM; idx += stride) {         // fc1T[384][96]
    int n = idx / DIM, k = idx - n * DIM;
    fc1T[idx] = (_Float16)fc1_w[k * HID + n];
  }
  for (int idx = i; idx < DIM * HID; idx += stride) {         // fc2T[96][384]
    int n = idx / HID, k = idx - n * HID;
    fc2T[idx] = (_Float16)fc2_w[k * DIM + n];
  }
}

// ---------------- attention: one block per 7x7 window ----------------
// dyn LDS layout (bytes):
//  xf  f32 [49][96]   @ 0      (18816)  shortcut, pre-LN
//  xw  f16 [64][96]   @ 18816  (12288)  LN out, later attn output o
//  qb  f16 [64][96]   @ 31104  (12288)
//  kb  f16 [64][96]   @ 43392  (12288)
//  vT  f16 [96][64]   @ 55680  (12288)
//  Sb  f32 [64][64]   @ 67968  (16384)
//  Pb  f16 [64][64]   @ 84352  (8192)   total 92544
#define ATTN_SMEM 92544

__global__ __launch_bounds__(256) void attn_kernel(
    const float* __restrict__ x,
    const float* __restrict__ n1w, const float* __restrict__ n1b,
    const _Float16* __restrict__ qkv_wT, const float* __restrict__ qkv_b,
    const _Float16* __restrict__ proj_wT, const float* __restrict__ proj_b,
    float* __restrict__ tout) {
  extern __shared__ char smem[];
  float*    xf = (float*)(smem);
  _Float16* xw = (_Float16*)(smem + 18816);
  _Float16* qb = (_Float16*)(smem + 31104);
  _Float16* kb = (_Float16*)(smem + 43392);
  _Float16* vT = (_Float16*)(smem + 55680);
  float*    Sb = (float*)(smem + 67968);
  _Float16* Pb = (_Float16*)(smem + 84352);

  const int tid  = threadIdx.x;
  const int lane = tid & 31;
  const int wave = tid >> 5;
  const int row4 = tid >> 2;     // 4 lanes cooperate per row
  const int sub4 = tid & 3;

  const int wi = blockIdx.x;
  const int b  = wi >> 10;          // 32*32 windows per image
  const int wh = (wi >> 5) & 31;
  const int ww = wi & 31;
  const int h0 = wh * WINSZ, w0 = ww * WINSZ;

  // ---- load window (c-major so each 7-token run is contiguous in W) ----
  for (int idx = tid; idx < NTOK * DIM; idx += 256) {
    int c  = idx / NTOK;
    int tk = idx - c * NTOK;
    int r = tk / WINSZ, cc = tk - r * WINSZ;
    size_t g = (((size_t)b * DIM + c) * HGT + (h0 + r)) * WID + (w0 + cc);
    xf[tk * DIM + c] = x[g];
  }
  for (int i = tid; i < (C3 * DIM * 2) / 64; i += 256)
    __builtin_prefetch((const char*)qkv_wT + i * 64, 0, 0);
  __syncthreads();

  // ---- LayerNorm1 -> xw (f16), 4 lanes per row; zero pad rows 49..63 ----
  if (row4 < NTOK) {
    const float* rowp = xf + row4 * DIM;
    float s = 0.f;
    for (int c = sub4; c < DIM; c += 4) s += rowp[c];
    float mu = red4_sum(s) * (1.0f / DIM);
    float v = 0.f;
    for (int c = sub4; c < DIM; c += 4) { float d = rowp[c] - mu; v += d * d; }
    float ri = rsqrtf(red4_sum(v) * (1.0f / DIM) + 1e-5f);
    _Float16* o = xw + row4 * DIM;
    for (int c = sub4; c < DIM; c += 4)
      o[c] = (_Float16)(((rowp[c] - mu) * ri) * n1w[c] + n1b[c]);
  } else {
    _Float16* o = xw + row4 * DIM;
    for (int c = sub4; c < DIM; c += 4) o[c] = (_Float16)0.f;
  }
  __syncthreads();

  // ---- QKV GEMM: [64x96] @ [96x288], 4x18 tiles, K = 3 steps ----
  for (int t = wave; t < 72; t += 8) {
    int mt = t / 18, nt = t - mt * 18;
    v8f acc = {};
#pragma unroll
    for (int ks = 0; ks < 3; ++ks) {
      F16Frag a, bf;
      load_a(a, xw, DIM, mt * 16, ks * 32, lane);
      load_b(bf, qkv_wT, DIM, ks * 32, nt * 16, lane);
      acc = wmma16(a, bf, acc);
    }
    const int n  = nt * 16 + (lane & 15);
    const int hf = lane >> 4;
    const float bias = qkv_b[n];
    // nt is wave-uniform and 16-aligned vs the 96/192 split -> scalar branch
    if (nt < 6) {
#pragma unroll
      for (int r = 0; r < 8; ++r)
        qb[(mt * 16 + r + 8 * hf) * DIM + n] = (_Float16)(acc[r] + bias);
    } else if (nt < 12) {
#pragma unroll
      for (int r = 0; r < 8; ++r)
        kb[(mt * 16 + r + 8 * hf) * DIM + (n - DIM)] = (_Float16)(acc[r] + bias);
    } else {
#pragma unroll
      for (int r = 0; r < 8; ++r)
        vT[(n - 2 * DIM) * MP + (mt * 16 + r + 8 * hf)] = (_Float16)(acc[r] + bias);
    }
  }
  __syncthreads();

  // ---- per-head attention ----
  const float scale = 0.17677669529663687f; // 1/sqrt(32)
  for (int hh = 0; hh < HEADS; ++hh) {
    // S = q @ k^T  (K = hd = 32, single WMMA step); 4x4 tiles
    for (int t = wave; t < 16; t += 8) {
      int mt = t >> 2, nt = t & 3;
      F16Frag a, bf;
      load_a(a, qb, DIM, mt * 16, hh * HD, lane);
      load_b(bf, kb, DIM, hh * HD, nt * 16, lane);  // kb row-major == B N-major
      v8f acc = {};
      acc = wmma16(a, bf, acc);
      const int n  = nt * 16 + (lane & 15);
      const int hf = lane >> 4;
#pragma unroll
      for (int r = 0; r < 8; ++r)
        Sb[(mt * 16 + r + 8 * hf) * MP + n] = acc[r] * scale;
    }
    __syncthreads();
    // softmax, 4 lanes per row; pad rows/cols -> 0
    {
      _Float16* pr = Pb + row4 * MP;
      if (row4 < NTOK) {
        const float* sr = Sb + row4 * MP;
        float mx = -3.4e38f;
        for (int c = sub4; c < NTOK; c += 4) mx = fmaxf(mx, sr[c]);
        mx = red4_max(mx);
        float sum = 0.f;
        for (int c = sub4; c < NTOK; c += 4) sum += __expf(sr[c] - mx);
        float inv = 1.0f / red4_sum(sum);
        for (int c = sub4; c < NTOK; c += 4)
          pr[c] = (_Float16)(__expf(sr[c] - mx) * inv);
        for (int c = NTOK + sub4; c < MP; c += 4) pr[c] = (_Float16)0.f;
      } else {
        for (int c = sub4; c < MP; c += 4) pr[c] = (_Float16)0.f;
      }
    }
    __syncthreads();
    // o = P @ v : 4x2 tiles, K = 64 (2 steps); one tile per wave
    {
      int mt = wave >> 1, nt = wave & 1;
      v8f acc = {};
#pragma unroll
      for (int ks = 0; ks < 2; ++ks) {
        F16Frag a, bf;
        load_a(a, Pb, MP, mt * 16, ks * 32, lane);
        load_b(bf, vT, MP, ks * 32, hh * HD + nt * 16, lane);
        acc = wmma16(a, bf, acc);
      }
      const int n  = hh * HD + nt * 16 + (lane & 15);
      const int hf = lane >> 4;
#pragma unroll
      for (int r = 0; r < 8; ++r)
        xw[(mt * 16 + r + 8 * hf) * DIM + n] = (_Float16)acc[r];  // reuse xw as o
    }
    __syncthreads();
  }

  // ---- proj + residual, write t[B][HW][C] ----
  for (int t = wave; t < 24; t += 8) {
    int mt = t / 6, nt = t - mt * 6;
    v8f acc = {};
#pragma unroll
    for (int ks = 0; ks < 3; ++ks) {
      F16Frag a, bf;
      load_a(a, xw, DIM, mt * 16, ks * 32, lane);
      load_b(bf, proj_wT, DIM, ks * 32, nt * 16, lane);
      acc = wmma16(a, bf, acc);
    }
    const int n  = nt * 16 + (lane & 15);
    const int hf = lane >> 4;
    const float bias = proj_b[n];
#pragma unroll
    for (int r = 0; r < 8; ++r) {
      int m = mt * 16 + r + 8 * hf;
      if (m < NTOK) {
        float val = acc[r] + bias + xf[m * DIM + n];
        int hw = (h0 + m / WINSZ) * WID + (w0 + m % WINSZ);
        tout[((size_t)b * HWCNT + hw) * DIM + n] = val;
      }
    }
  }
}

// ---------------- MLP: one block per 64 tokens ----------------
// dyn LDS: tb f32 [64][96] @0 (24576) | ab f16 [64][96] @24576 (12288)
//          hb f16 [64][384] @36864 (49152)  total 86016
#define MLP_SMEM 86016

__global__ __launch_bounds__(256) void mlp_kernel(
    const float* __restrict__ tin,
    const float* __restrict__ n2w, const float* __restrict__ n2b,
    const _Float16* __restrict__ fc1_wT, const float* __restrict__ fc1_b,
    const _Float16* __restrict__ fc2_wT, const float* __restrict__ fc2_b,
    float* __restrict__ out) {
  extern __shared__ char smem[];
  float*    tb = (float*)smem;
  _Float16* ab = (_Float16*)(smem + 24576);
  _Float16* hb = (_Float16*)(smem + 36864);

  const int tid = threadIdx.x, lane = tid & 31, wave = tid >> 5;
  const int row4 = tid >> 2, sub4 = tid & 3;
  const size_t g0 = (size_t)blockIdx.x * MP;      // first token of tile
  const int b   = (int)(g0 / HWCNT);              // tiles never cross batch (50176 % 64 == 0)
  const int hw0 = (int)(g0 - (size_t)b * HWCNT);

  const float* src = tin + g0 * DIM;
#ifdef HAVE_ASYNC_LDS
  {
    // builtin expects generic int4* args (per compiler diagnostic)
    v4i* gsrc = (v4i*)src;
    v4i* ldst = (v4i*)tb;
    for (int i = tid; i < (MP * DIM) / 4; i += 256)   // 16B per thread-iter
      __builtin_amdgcn_global_load_async_to_lds_b128(gsrc + i, ldst + i, 0, 0);
  }
#else
  for (int i = tid; i < MP * DIM; i += 256) tb[i] = src[i];
#endif
  for (int i = tid; i < (DIM * HID * 2) / 64; i += 256)
    __builtin_prefetch((const char*)fc1_wT + i * 64, 0, 0);
#ifdef HAVE_ASYNC_LDS
#if __has_builtin(__builtin_amdgcn_s_wait_asynccnt)
  __builtin_amdgcn_s_wait_asynccnt(0);
#else
  asm volatile("s_wait_asynccnt 0x0" ::: "memory");
#endif
#endif
  __syncthreads();

  // LayerNorm2 -> ab (f16), 4 lanes per row (64 rows x 4 = 256 threads)
  {
    const float* rowp = tb + row4 * DIM;
    float s = 0.f;
    for (int c = sub4; c < DIM; c += 4) s += rowp[c];
    float mu = red4_sum(s) * (1.0f / DIM);
    float v = 0.f;
    for (int c = sub4; c < DIM; c += 4) { float d = rowp[c] - mu; v += d * d; }
    float ri = rsqrtf(red4_sum(v) * (1.0f / DIM) + 1e-5f);
    _Float16* o = ab + row4 * DIM;
    for (int c = sub4; c < DIM; c += 4)
      o[c] = (_Float16)(((rowp[c] - mu) * ri) * n2w[c] + n2b[c]);
  }
  __syncthreads();

  // fc1 + exact GELU: [64x96]@[96x384], 4x24 tiles, K = 3 steps
  for (int t = wave; t < 96; t += 8) {
    int mt = t / 24, nt = t - mt * 24;
    v8f acc = {};
#pragma unroll
    for (int ks = 0; ks < 3; ++ks) {
      F16Frag a, bf;
      load_a(a, ab, DIM, mt * 16, ks * 32, lane);
      load_b(bf, fc1_wT, DIM, ks * 32, nt * 16, lane);
      acc = wmma16(a, bf, acc);
    }
    const int n  = nt * 16 + (lane & 15);
    const int hf = lane >> 4;
    const float bias = fc1_b[n];
#pragma unroll
    for (int r = 0; r < 8; ++r) {
      float v = acc[r] + bias;
      float g = 0.5f * v * (1.0f + erff(v * 0.70710678118654752f));
      hb[(mt * 16 + r + 8 * hf) * HID + n] = (_Float16)g;
    }
  }
  __syncthreads();

  // fc2 + residual (in place into tb): 4x6 tiles, K = 12 steps
  for (int t = wave; t < 24; t += 8) {
    int mt = t / 6, nt = t - mt * 6;
    v8f acc = {};
#pragma unroll
    for (int ks = 0; ks < 12; ++ks) {
      F16Frag a, bf;
      load_a(a, hb, HID, mt * 16, ks * 32, lane);
      load_b(bf, fc2_wT, HID, ks * 32, nt * 16, lane);
      acc = wmma16(a, bf, acc);
    }
    const int n  = nt * 16 + (lane & 15);
    const int hf = lane >> 4;
    const float bias = fc2_b[n];
#pragma unroll
    for (int r = 0; r < 8; ++r) {
      int m = mt * 16 + r + 8 * hf;
      tb[m * DIM + n] = tb[m * DIM + n] + acc[r] + bias;
    }
  }
  __syncthreads();

  // coalesced transposed writeout: out[b][c][hw0+m]
  for (int i = tid; i < DIM * MP; i += 256) {
    int c = i >> 6, m = i & 63;
    out[((size_t)b * DIM + c) * HWCNT + hw0 + m] = tb[m * DIM + c];
  }
}

extern "C" void kernel_launch(void* const* d_in, const int* in_sizes, int n_in,
                              void* d_out, int out_size, void* d_ws, size_t ws_size,
                              hipStream_t stream) {
  (void)in_sizes; (void)n_in; (void)out_size; (void)ws_size;
  const float* x      = (const float*)d_in[0];
  const float* n1w    = (const float*)d_in[1];
  const float* n1b    = (const float*)d_in[2];
  const float* qkv_w  = (const float*)d_in[3];
  const float* qkv_b  = (const float*)d_in[4];
  const float* proj_w = (const float*)d_in[5];
  const float* proj_b = (const float*)d_in[6];
  const float* n2w    = (const float*)d_in[7];
  const float* n2b    = (const float*)d_in[8];
  const float* fc1_w  = (const float*)d_in[9];
  const float* fc1_b  = (const float*)d_in[10];
  const float* fc2_w  = (const float*)d_in[11];
  const float* fc2_b  = (const float*)d_in[12];

  // workspace layout
  char* ws = (char*)d_ws;
  float* tbuf = (float*)ws;                                  // [8][50176][96] f32
  size_t off = (size_t)BATCH * HWCNT * DIM * sizeof(float);  // 154,140,672 (256-aligned)
  _Float16* qkvT = (_Float16*)(ws + off); off += (size_t)C3 * DIM * 2;
  _Float16* projT = (_Float16*)(ws + off); off += (size_t)DIM * DIM * 2;
  _Float16* fc1T = (_Float16*)(ws + off); off += (size_t)HID * DIM * 2;
  _Float16* fc2T = (_Float16*)(ws + off); off += (size_t)DIM * HID * 2;

  prep_kernel<<<128, 256, 0, stream>>>(qkv_w, proj_w, fc1_w, fc2_w,
                                       qkvT, projT, fc1T, fc2T);
  attn_kernel<<<NWIN, 256, ATTN_SMEM, stream>>>(x, n1w, n1b, qkvT, qkv_b,
                                                projT, proj_b, tbuf);
  mlp_kernel<<<(BATCH * HWCNT) / MP, 256, MLP_SMEM, stream>>>(
      tbuf, n2w, n2b, fc1T, fc1_b, fc2T, fc2_b, (float*)d_out);
}
